// DeltaNet_31877247271564
// MI455X (gfx1250) — compile-verified, matrix operand order checked by reading, not run on hardware
//
#include <hip/hip_runtime.h>
#include <hip/hip_bf16.h>
#include <stdint.h>

typedef __bf16 bf16_t;
typedef __attribute__((ext_vector_type(16))) __bf16 v16bf;
typedef __attribute__((ext_vector_type(8)))  float  v8f;

#define B_   2
#define L_   4096
#define D_   1024
#define H_   4
#define HD_  256
#define CH_  128   // chunks per sequence
#define CS_  32    // chunk size

// ---------- bf16 helpers (RNE) ----------
__device__ __forceinline__ unsigned short f2bfbits(float f){
    unsigned u = __builtin_bit_cast(unsigned, f);
    unsigned r = u + 0x7fffu + ((u >> 16) & 1u);
    return (unsigned short)(r >> 16);
}
__device__ __forceinline__ bf16_t f2bf(float f){
    unsigned short s = f2bfbits(f);
    return __builtin_bit_cast(bf16_t, s);
}
__device__ __forceinline__ float bfbits2f(unsigned short u){
    unsigned x = ((unsigned)u) << 16;
    return __builtin_bit_cast(float, x);
}
__device__ __forceinline__ float bf2f(bf16_t v){ return (float)v; }
__device__ __forceinline__ unsigned pack2(float a, float b){
    return (unsigned)f2bfbits(a) | ((unsigned)f2bfbits(b) << 16);
}
__device__ __forceinline__ float sigm(float x){ return 1.f / (1.f + __expf(-x)); }

union FragB { v16bf v; uint4 q[2]; };

__device__ __forceinline__ v8f wmma_bf16(v16bf a, v16bf b, v8f c){
    // (neg_a, A, neg_b, B, c_mod, C, reuse_a, reuse_b)
    return __builtin_amdgcn_wmma_f32_16x16x32_bf16(false, a, false, b, (short)0, c, false, false);
}
__device__ __forceinline__ uint4 packv8(v8f v){
    uint4 r;
    r.x = pack2(v[0], v[1]); r.y = pack2(v[2], v[3]);
    r.z = pack2(v[4], v[5]); r.w = pack2(v[6], v[7]);
    return r;
}
__device__ __forceinline__ v8f unpackv8_scaled(uint4 u, float g){
    v8f c;
    unsigned wds[4] = {u.x, u.y, u.z, u.w};
    #pragma unroll
    for(int i = 0; i < 4; ++i){
        c[2*i]   = bfbits2f((unsigned short)(wds[i] & 0xffffu)) * g;
        c[2*i+1] = bfbits2f((unsigned short)(wds[i] >> 16)) * g;
    }
    return c;
}

// ---------- CDNA5 async global->LDS DMA (ASYNCcnt path) ----------
__device__ __forceinline__ void async_g2l_b128(unsigned lds_off, const void* gaddr){
    // per-lane: 16B  global -> LDS[lds_off], tracked by ASYNCcnt, bypasses VGPRs
    asm volatile("global_load_async_to_lds_b128 %0, %1, off"
                 :: "v"(lds_off), "v"(gaddr) : "memory");
}
__device__ __forceinline__ void wait_async0(){
    asm volatile("s_wait_asynccnt 0x0" ::: "memory");
}

// ---------- f32 -> bf16 conversion ----------
__global__ void __launch_bounds__(256) cvt_f32_bf16_kernel(const float* __restrict__ src,
                                                           bf16_t* __restrict__ dst, int n){
    int i = blockIdx.x * 256 + threadIdx.x;
    if(i < n) dst[i] = f2bf(src[i]);
}

// ---------- bf16 WMMA GEMM: C[M,N] = A[M,K] @ B[K,N] ----------
template<bool OUT_BF16>
__global__ void __launch_bounds__(256) gemm_bf16_kernel(const bf16_t* __restrict__ A,
                                                        const bf16_t* __restrict__ Bm,
                                                        void* __restrict__ Cout,
                                                        int M, int N, int K){
    __shared__ bf16_t As[128 * 40];   // A tile 128x32, stride 40 (bank-conflict pad)
    __shared__ bf16_t Bts[128 * 40];  // B tile transposed: [n][k]
    const int tid = threadIdx.x;
    const int wave = tid >> 5, lane = tid & 31, l16 = lane & 15, hi = (lane >> 4) & 1;
    const int wx = wave >> 2, wy = wave & 3;   // 2 M-groups x 4 N-groups of waves
    const int m0 = blockIdx.y * 128, n0 = blockIdx.x * 128;
    const unsigned asbase = (unsigned)(uintptr_t)(void*)As;   // LDS byte offset of As

    v8f acc[4][2] = {};
    for(int k0 = 0; k0 < K; k0 += 32){
        { // A tile: async DMA straight into (padded) LDS, 2x16B per lane
            int row = tid >> 1, half = tid & 1;
            const bf16_t* gp = A + (size_t)(m0 + row) * K + k0 + half * 16;
            unsigned lo = asbase + (unsigned)(row * 80 + half * 32);
            async_g2l_b128(lo,      gp);
            async_g2l_b128(lo + 16, gp + 8);
        }
        { // B tile: coalesced 32B read, transposed scatter to LDS
            int kk = tid >> 3, seg = tid & 7;
            const uint4* gq = (const uint4*)(Bm + (size_t)(k0 + kk) * N + n0 + seg * 16);
            uint4 v0 = gq[0], v1 = gq[1];
            unsigned wds[8] = {v0.x, v0.y, v0.z, v0.w, v1.x, v1.y, v1.z, v1.w};
            #pragma unroll
            for(int i = 0; i < 8; ++i){
                unsigned w2 = wds[i];
                Bts[(seg*16 + 2*i    ) * 40 + kk] = __builtin_bit_cast(bf16_t, (unsigned short)(w2 & 0xffffu));
                Bts[(seg*16 + 2*i + 1) * 40 + kk] = __builtin_bit_cast(bf16_t, (unsigned short)(w2 >> 16));
            }
        }
        wait_async0();          // this wave's async copies landed in LDS
        __syncthreads();        // all waves' copies visible
        FragB af[4], bfr[2];
        #pragma unroll
        for(int i = 0; i < 4; ++i){
            const bf16_t* p = As + (wx*64 + i*16 + l16) * 40 + hi * 8;
            af[i].q[0] = *(const uint4*)p;
            af[i].q[1] = *(const uint4*)(p + 16);
        }
        #pragma unroll
        for(int j = 0; j < 2; ++j){
            const bf16_t* p = Bts + (wy*32 + j*16 + l16) * 40 + hi * 8;
            bfr[j].q[0] = *(const uint4*)p;
            bfr[j].q[1] = *(const uint4*)(p + 16);
        }
        #pragma unroll
        for(int i = 0; i < 4; ++i)
            #pragma unroll
            for(int j = 0; j < 2; ++j)
                acc[i][j] = wmma_bf16(af[i].v, bfr[j].v, acc[i][j]);
        __syncthreads();
    }
    // epilogue (branch-free per specialization)
    #pragma unroll
    for(int i = 0; i < 4; ++i){
        #pragma unroll
        for(int j = 0; j < 2; ++j){
            int col = n0 + wy*32 + j*16 + l16;
            int mb  = m0 + wx*64 + i*16 + hi*8;
            #pragma unroll
            for(int r = 0; r < 8; ++r){
                float v = acc[i][j][r];
                size_t idx = (size_t)(mb + r) * N + col;
                if constexpr (OUT_BF16) ((bf16_t*)Cout)[idx] = f2bf(v);
                else                    ((float*)Cout)[idx]  = v;
            }
        }
    }
}

// ---------- beta/gamma = sigmoid(x @ Wb / Wgamma) ----------
__global__ void __launch_bounds__(256) betagamma_kernel(const float* __restrict__ x,
                                                        const float* __restrict__ Wb,
                                                        const float* __restrict__ Wg,
                                                        float* __restrict__ beta,
                                                        float* __restrict__ gamma){
    __shared__ float red[256];
    int bl = blockIdx.x, tid = threadIdx.x;
    int grp = tid >> 5, lane = tid & 31;
    const float* W = (grp < 4) ? Wb : Wg;
    int col = grp & 3;
    const float* xr = x + (size_t)bl * D_;
    float p = 0.f;
    for(int j = lane; j < D_; j += 32) p += xr[j] * W[j * H_ + col];
    red[tid] = p; __syncthreads();
    for(int s = 16; s > 0; s >>= 1){ if(lane < s) red[tid] += red[tid + s]; __syncthreads(); }
    if(lane == 0){
        float v = sigm(red[tid]);
        if(grp < 4) beta[(size_t)bl * H_ + col] = v;
        else        gamma[(size_t)bl * H_ + col] = v;
    }
}

// ---------- prep: conv+SiLU + rotary mix + l2norm + UT-transform; emits WMMA layouts ----------
__global__ void __launch_bounds__(256) prep_kernel(
    const bf16_t* __restrict__ qf, const bf16_t* __restrict__ kf, const bf16_t* __restrict__ vf,
    const float* __restrict__ cq, const float* __restrict__ ck, const float* __restrict__ cv,
    const float* __restrict__ rot, const float* __restrict__ beta, const float* __restrict__ gamma,
    bf16_t* __restrict__ qn, bf16_t* __restrict__ wout, bf16_t* __restrict__ kT,
    bf16_t* __restrict__ attn, float* __restrict__ uT, float* __restrict__ gm)
{
    extern __shared__ char smem[];
    float* qL  = (float*)(smem);            // 32x256 f32
    float* kL  = (float*)(smem + 32768);
    float* kbL = (float*)(smem + 65536);
    float* vL  = (float*)(smem + 98304);
    float* Am  = (float*)(smem + 131072);   // 32x32
    float* red = (float*)(smem + 135168);   // 256

    int blk = blockIdx.x;
    int bh = blk >> 7, cn = blk & 127;
    int b = bh >> 2, h = bh & 3;
    int t = threadIdx.x;
    float mix = sigm(rot[h]);
    int jh = t & 127;
    float invf = __expf(-(float)jh * (9.210340371976184f / 128.f));  // 10000^(-2j/256)
    size_t cb = (size_t)bh * CH_ + cn;

    for(int r = 0; r < CS_; ++r){
        int l = cn * CS_ + r;
        size_t bl = (size_t)b * L_ + l;
        auto conv = [&](const bf16_t* yf, const float* cw, int c) -> float {
            float a = 0.f;
            #pragma unroll
            for(int jj = 0; jj < 4; ++jj){
                int ls = l + jj - 3;
                if(ls >= 0)
                    a += bf2f(yf[((size_t)b * L_ + ls) * D_ + h * HD_ + c]) * cw[(h * HD_ + c) * 4 + jj];
            }
            return a * sigm(a);  // SiLU
        };
        float q0 = conv(qf, cq, t), q1 = conv(qf, cq, 2*jh), q2 = conv(qf, cq, 2*jh + 1);
        float k0 = conv(kf, ck, t), k1 = conv(kf, ck, 2*jh), k2 = conv(kf, ck, 2*jh + 1);
        float vv = conv(vf, cv, t);
        float ang = (float)l * invf;
        float sn = __sinf(ang), cs = __cosf(ang);
        float qr = (t < 128) ? (q1*cs - q2*sn) : (q1*sn + q2*cs);
        float kr = (t < 128) ? (k1*cs - k2*sn) : (k1*sn + k2*cs);
        float qv = mix*qr + (1.f - mix)*q0;
        float kv = mix*kr + (1.f - mix)*k0;
        red[t] = qv*qv; __syncthreads();
        for(int s = 128; s > 0; s >>= 1){ if(t < s) red[t] += red[t + s]; __syncthreads(); }
        float qss = red[0]; __syncthreads();
        red[t] = kv*kv; __syncthreads();
        for(int s = 128; s > 0; s >>= 1){ if(t < s) red[t] += red[t + s]; __syncthreads(); }
        float kss = red[0]; __syncthreads();
        float qnv = qv * rsqrtf(qss + 1e-6f);
        float knv = kv * rsqrtf(kss + 1e-6f);
        float be  = beta[bl * H_ + h];
        qL[r*256 + t] = qnv; kL[r*256 + t] = knv;
        kbL[r*256 + t] = knv * be; vL[r*256 + t] = vv * be;
        qn[(cb * CS_ + r) * HD_ + t] = f2bf(qnv);          // q  row-major (A-operand)
        kT[(cb * HD_ + t) * CS_ + r] = f2bf(knv);          // k^T row-major (A-operand)
        __syncthreads();
    }
    if(t == 0){
        float gs = 0.f;
        for(int r = 0; r < CS_; ++r) gs += gamma[((size_t)b * L_ + cn * CS_ + r) * H_ + h];
        gm[cb] = gs * (1.f / 32.f);
    }
    // A = -(kb @ k^T) * strict_lower
    for(int e = t; e < 1024; e += 256){
        int i = e >> 5, jj = e & 31;
        float a = 0.f;
        if(i > jj){ for(int d = 0; d < HD_; ++d) a += kbL[i*256 + d] * kL[jj*256 + d]; a = -a; }
        Am[e] = a;
    }
    __syncthreads();
    // UT-transform forward recursion (serial in i, parallel in columns)
    for(int i = 1; i < CS_; ++i){
        float nv = 0.f;
        if(t < i){
            nv = Am[i*32 + t];
            for(int d = 0; d < i; ++d) nv += Am[i*32 + d] * Am[d*32 + t];
        }
        __syncthreads();
        if(t < i) Am[i*32 + t] = nv;
        __syncthreads();
    }
    // u = T@v (f32, transposed layout), w = T@kb (bf16 row-major)
    for(int r = 0; r < CS_; ++r){
        float su = vL[r*256 + t];
        float sw = kbL[r*256 + t];
        for(int d = 0; d < r; ++d){ float a = Am[r*32 + d]; su += a * vL[d*256 + t]; sw += a * kbL[d*256 + t]; }
        uT[(cb * HD_ + t) * CS_ + r] = su;
        wout[(cb * CS_ + r) * HD_ + t] = f2bf(sw);
    }
    // attn = (q k^T) * tril (incl diag), bf16 row-major (A-operand)
    for(int e = t; e < 1024; e += 256){
        int i = e >> 5, jj = e & 31;
        float a = 0.f;
        if(i >= jj){ for(int d = 0; d < HD_; ++d) a += qL[i*256 + d] * kL[jj*256 + d]; }
        attn[cb * 1024 + e] = f2bf(a);
    }
}

// ---------- sequential chunk scan: S lives in LDS (bf16, padded); all O(n^3) on WMMA ----------
__global__ void __launch_bounds__(256) scan_kernel(
    const bf16_t* __restrict__ qn, const bf16_t* __restrict__ wm,
    const bf16_t* __restrict__ kTg, const bf16_t* __restrict__ at,
    const float* __restrict__ uT, const float* __restrict__ gm,
    float* __restrict__ of)
{
    extern __shared__ char smem[];
    bf16_t* St  = (bf16_t*)smem;                    // 256 x 264 (S transposed: [dv][dk])
    bf16_t* utT = (bf16_t*)(smem + 256 * 264 * 2);  // 256 x 40  (u_t transposed: [dv][chunk])
    const int STS = 264, UTS = 40;
    int tid = threadIdx.x, wave = tid >> 5, lane = tid & 31, l16 = lane & 15, hi = (lane >> 4) & 1;
    int bh = blockIdx.x, b = bh >> 2, h = bh & 3;

    for(int i = tid; i < 256 * STS; i += 256) St[i] = __builtin_bit_cast(bf16_t, (unsigned short)0);
    __syncthreads();

    for(int cn = 0; cn < CH_; ++cn){
        size_t base = (size_t)bh * CH_ + cn;
        const bf16_t* qp = qn  + base * CS_ * HD_;
        const bf16_t* wp = wm  + base * CS_ * HD_;
        const bf16_t* kp = kTg + base * HD_ * CS_;
        const bf16_t* ap = at  + base * CS_ * CS_;
        const float*  up = uT  + base * HD_ * CS_;
        float g = gm[base];
        if(cn + 1 < CH_){
            __builtin_prefetch(qp + CS_ * HD_, 0, 0);
            __builtin_prefetch(wp + CS_ * HD_, 0, 0);
            __builtin_prefetch(kp + HD_ * CS_, 0, 0);
        }
        // ---- step 1: u_t = u - w @ S  ->  utT (bf16)
        {
            v8f acc[2][2] = {};
            for(int ks = 0; ks < 8; ++ks){
                int kk0 = ks * 32;
                FragB af[2], bfr[2];
                #pragma unroll
                for(int mt = 0; mt < 2; ++mt){
                    const bf16_t* p = wp + (size_t)(mt*16 + l16) * HD_ + kk0 + hi * 8;
                    af[mt].q[0] = *(const uint4*)p; af[mt].q[1] = *(const uint4*)(p + 16);
                }
                #pragma unroll
                for(int jn = 0; jn < 2; ++jn){
                    const bf16_t* p = St + (size_t)((2*wave + jn)*16 + l16) * STS + kk0 + hi * 8;
                    bfr[jn].q[0] = *(const uint4*)p; bfr[jn].q[1] = *(const uint4*)(p + 16);
                }
                #pragma unroll
                for(int mt = 0; mt < 2; ++mt)
                    #pragma unroll
                    for(int jn = 0; jn < 2; ++jn)
                        acc[mt][jn] = wmma_bf16(af[mt].v, bfr[jn].v, acc[mt][jn]);
            }
            #pragma unroll
            for(int mt = 0; mt < 2; ++mt)
                #pragma unroll
                for(int jn = 0; jn < 2; ++jn){
                    int n  = (2*wave + jn)*16 + l16;
                    int m0 = mt*16 + hi*8;
                    const float* pu = up + (size_t)n * CS_ + m0;
                    v8f ut;
                    #pragma unroll
                    for(int r = 0; r < 8; ++r) ut[r] = pu[r] - acc[mt][jn][r];
                    *(uint4*)(utT + (size_t)n * UTS + m0) = packv8(ut);
                }
        }
        __syncthreads();
        // ---- step 2: o = q @ S + attn @ u_t
        {
            v8f acc[2][2] = {};
            for(int ks = 0; ks < 8; ++ks){
                int kk0 = ks * 32;
                FragB af[2], bfr[2];
                #pragma unroll
                for(int mt = 0; mt < 2; ++mt){
                    const bf16_t* p = qp + (size_t)(mt*16 + l16) * HD_ + kk0 + hi * 8;
                    af[mt].q[0] = *(const uint4*)p; af[mt].q[1] = *(const uint4*)(p + 16);
                }
                #pragma unroll
                for(int jn = 0; jn < 2; ++jn){
                    const bf16_t* p = St + (size_t)((2*wave + jn)*16 + l16) * STS + kk0 + hi * 8;
                    bfr[jn].q[0] = *(const uint4*)p; bfr[jn].q[1] = *(const uint4*)(p + 16);
                }
                #pragma unroll
                for(int mt = 0; mt < 2; ++mt)
                    #pragma unroll
                    for(int jn = 0; jn < 2; ++jn)
                        acc[mt][jn] = wmma_bf16(af[mt].v, bfr[jn].v, acc[mt][jn]);
            }
            FragB aa[2], bu[2];
            #pragma unroll
            for(int mt = 0; mt < 2; ++mt){
                const bf16_t* p = ap + (size_t)(mt*16 + l16) * CS_ + hi * 8;
                aa[mt].q[0] = *(const uint4*)p; aa[mt].q[1] = *(const uint4*)(p + 16);
            }
            #pragma unroll
            for(int jn = 0; jn < 2; ++jn){
                const bf16_t* p = utT + (size_t)((2*wave + jn)*16 + l16) * UTS + hi * 8;
                bu[jn].q[0] = *(const uint4*)p; bu[jn].q[1] = *(const uint4*)(p + 16);
            }
            #pragma unroll
            for(int mt = 0; mt < 2; ++mt)
                #pragma unroll
                for(int jn = 0; jn < 2; ++jn)
                    acc[mt][jn] = wmma_bf16(aa[mt].v, bu[jn].v, acc[mt][jn]);
            #pragma unroll
            for(int mt = 0; mt < 2; ++mt)
                #pragma unroll
                for(int jn = 0; jn < 2; ++jn){
                    int n  = (2*wave + jn)*16 + l16;
                    int m0 = mt*16 + hi*8;
                    #pragma unroll
                    for(int r = 0; r < 8; ++r){
                        int l = cn * CS_ + m0 + r;
                        of[((size_t)b * L_ + l) * D_ + h * HD_ + n] = acc[mt][jn][r];
                    }
                }
        }
        __syncthreads();
        // ---- step 3: S = g*S + k^T @ u_t   (C-operand carries g*S_old)
        {
            #pragma unroll
            for(int m2 = 0; m2 < 2; ++m2){
                int mt = wave * 2 + m2;
                FragB ka;
                const bf16_t* p = kp + (size_t)(mt*16 + l16) * CS_ + hi * 8;
                ka.q[0] = *(const uint4*)p; ka.q[1] = *(const uint4*)(p + 16);
                for(int nt = 0; nt < 16; ++nt){
                    FragB bu;
                    const bf16_t* pb = utT + (size_t)(nt*16 + l16) * UTS + hi * 8;
                    bu.q[0] = *(const uint4*)pb; bu.q[1] = *(const uint4*)(pb + 16);
                    int n  = nt*16 + l16;
                    int m0 = mt*16 + hi*8;
                    bf16_t* ps = St + (size_t)n * STS + m0;
                    v8f c = unpackv8_scaled(*(const uint4*)ps, g);
                    v8f d = wmma_bf16(ka.v, bu.v, c);
                    *(uint4*)ps = packv8(d);
                }
            }
        }
        __syncthreads();
    }
}

// ---------- per-head RMSNorm of delta output -> bf16 for Wo GEMM ----------
__global__ void __launch_bounds__(256) rmsnorm_o_kernel(const float* __restrict__ of,
                                                        const float* __restrict__ onw,
                                                        bf16_t* __restrict__ ob){
    __shared__ float red[256];
    int bl = blockIdx.x, t = threadIdx.x;
    int h = t >> 6, i = t & 63;
    size_t base = (size_t)bl * D_ + h * HD_;
    float vs[4]; float p = 0.f;
    #pragma unroll
    for(int m = 0; m < 4; ++m){ float v = of[base + i + 64*m]; vs[m] = v; p += v*v; }
    red[t] = p; __syncthreads();
    for(int s = 32; s > 0; s >>= 1){ if(i < s) red[t] += red[t + s]; __syncthreads(); }
    float scale = rsqrtf(red[h * 64] / 256.f + 1e-5f);
    #pragma unroll
    for(int m = 0; m < 4; ++m){
        int n = i + 64*m;
        ob[base + n] = f2bf(vs[m] * scale * onw[n]);
    }
}

// ---------- fractal mixer + rmsnorm + gate + final sum ----------
__global__ void __launch_bounds__(256) final_kernel(const float* __restrict__ x,
                                                    const float* __restrict__ core,
                                                    const float* __restrict__ fcw,
                                                    const float* __restrict__ fgw,
                                                    const float* __restrict__ fgb,
                                                    const float* __restrict__ mnw,
                                                    float* __restrict__ out){
    __shared__ float red[256];
    int bl = blockIdx.x, t = threadIdx.x;
    int l = bl & (L_ - 1);
    size_t base = (size_t)bl * D_;
    float fr[4]; float ssq = 0.f, gp = 0.f;
    #pragma unroll
    for(int m = 0; m < 4; ++m){
        int c = t + 256*m;
        float xv = x[base + c];
        float o = 2.f * xv;
        #pragma unroll
        for(int i = 0; i < 4; ++i){
            int dil = 1 << i;
            float xs = (l >= dil) ? x[base - (size_t)dil * D_ + c] : 0.f;
            o += xs * fcw[(i*D_ + c)*2] + xv * fcw[(i*D_ + c)*2 + 1];
        }
        fr[m] = o; ssq += o*o; gp += xv * fgw[c];
    }
    red[t] = ssq; __syncthreads();
    for(int s = 128; s > 0; s >>= 1){ if(t < s) red[t] += red[t + s]; __syncthreads(); }
    float tot = red[0]; __syncthreads();
    red[t] = gp; __syncthreads();
    for(int s = 128; s > 0; s >>= 1){ if(t < s) red[t] += red[t + s]; __syncthreads(); }
    float gd = red[0];
    float scale = rsqrtf(tot / (float)D_ + 1e-5f);
    float gate = sigm(gd + fgb[0]);
    #pragma unroll
    for(int m = 0; m < 4; ++m){
        int c = t + 256*m;
        out[base + c] = core[base + c] + gate * fr[m] * scale * mnw[c];
    }
}

extern "C" void kernel_launch(void* const* d_in, const int* in_sizes, int n_in,
                              void* d_out, int out_size, void* d_ws, size_t ws_size,
                              hipStream_t stream){
    const float* x   = (const float*)d_in[0];
    const float* Wq  = (const float*)d_in[1];
    const float* Wk  = (const float*)d_in[2];
    const float* Wv  = (const float*)d_in[3];
    const float* Wb  = (const float*)d_in[4];
    const float* Wg  = (const float*)d_in[5];
    const float* rot = (const float*)d_in[6];
    const float* cq  = (const float*)d_in[7];
    const float* ck  = (const float*)d_in[8];
    const float* cv  = (const float*)d_in[9];
    const float* onw = (const float*)d_in[10];
    const float* Wo  = (const float*)d_in[11];
    const float* fcw = (const float*)d_in[12];
    const float* fgw = (const float*)d_in[13];
    const float* fgb = (const float*)d_in[14];
    const float* mnw = (const float*)d_in[15];
    float* out = (float*)d_out;

    char* ws = (char*)d_ws;
    size_t off = 0;
    auto alloc = [&](size_t bytes) -> char* {
        char* p = ws + off;
        off = (off + bytes + 255) & ~(size_t)255;
        return p;
    };
    const size_t NBLD = (size_t)B_ * L_ * D_;
    bf16_t* xb   = (bf16_t*)alloc(NBLD * 2);
    bf16_t* Wqb  = (bf16_t*)alloc((size_t)D_ * D_ * 2);
    bf16_t* Wkb  = (bf16_t*)alloc((size_t)D_ * D_ * 2);
    bf16_t* Wvb  = (bf16_t*)alloc((size_t)D_ * D_ * 2);
    bf16_t* Wob  = (bf16_t*)alloc((size_t)D_ * D_ * 2);
    bf16_t* qfb  = (bf16_t*)alloc(NBLD * 2);
    bf16_t* kfb  = (bf16_t*)alloc(NBLD * 2);
    bf16_t* vfb  = (bf16_t*)alloc(NBLD * 2);
    float*  betaB  = (float*)alloc((size_t)B_ * L_ * H_ * 4);
    float*  gammaB = (float*)alloc((size_t)B_ * L_ * H_ * 4);
    bf16_t* qnb  = (bf16_t*)alloc(NBLD * 2);
    bf16_t* wmb  = (bf16_t*)alloc(NBLD * 2);
    bf16_t* kTb  = (bf16_t*)alloc(NBLD * 2);
    bf16_t* atb  = (bf16_t*)alloc((size_t)B_ * H_ * CH_ * CS_ * CS_ * 2);
    float*  uTf  = (float*)alloc(NBLD * 4);
    float*  gmf  = (float*)alloc((size_t)B_ * H_ * CH_ * 4);
    float*  of   = (float*)alloc(NBLD * 4);
    // safe aliases: consumed-before-overwritten within a single launch sequence
    bf16_t* obb   = qnb;            // qn fully consumed by scan before rmsnorm_o writes ob
    float*  coreF = uTf;            // uT fully consumed by scan before core GEMM writes

    const int nBLD = (int)NBLD, nDD = D_ * D_;
    cvt_f32_bf16_kernel<<<(nBLD + 255)/256, 256, 0, stream>>>(x,  xb,  nBLD);
    cvt_f32_bf16_kernel<<<(nDD  + 255)/256, 256, 0, stream>>>(Wq, Wqb, nDD);
    cvt_f32_bf16_kernel<<<(nDD  + 255)/256, 256, 0, stream>>>(Wk, Wkb, nDD);
    cvt_f32_bf16_kernel<<<(nDD  + 255)/256, 256, 0, stream>>>(Wv, Wvb, nDD);
    cvt_f32_bf16_kernel<<<(nDD  + 255)/256, 256, 0, stream>>>(Wo, Wob, nDD);

    dim3 gg(D_ / 128, (B_ * L_) / 128);   // (8, 64)
    gemm_bf16_kernel<true><<<gg, 256, 0, stream>>>(xb, Wqb, qfb, B_ * L_, D_, D_);
    gemm_bf16_kernel<true><<<gg, 256, 0, stream>>>(xb, Wkb, kfb, B_ * L_, D_, D_);
    gemm_bf16_kernel<true><<<gg, 256, 0, stream>>>(xb, Wvb, vfb, B_ * L_, D_, D_);

    betagamma_kernel<<<B_ * L_, 256, 0, stream>>>(x, Wb, Wg, betaB, gammaB);

    prep_kernel<<<B_ * H_ * CH_, 256, 136192, stream>>>(qfb, kfb, vfb, cq, ck, cv, rot,
                                                        betaB, gammaB,
                                                        qnb, wmb, kTb, atb, uTf, gmf);

    scan_kernel<<<B_ * H_, 256, 155648, stream>>>(qnb, wmb, kTb, atb, uTf, gmf, of);

    rmsnorm_o_kernel<<<B_ * L_, 256, 0, stream>>>(of, onw, obb);

    gemm_bf16_kernel<false><<<gg, 256, 0, stream>>>(obb, Wob, coreF, B_ * L_, D_, D_);

    final_kernel<<<B_ * L_, 256, 0, stream>>>(x, coreF, fcw, fgw, fgb, mnw, out);
}
